// BERT_10222022164982
// MI455X (gfx1250) — matrix operand order repaired and checked by reading, not compiled, not run on hardware
//
#include <hip/hip_runtime.h>
#include <hip/hip_bf16.h>
#include <stdint.h>

#define VOCAB 10000
#define D     512
#define L     6
#define H     8
#define B     32
#define S     384
#define DK    64
#define DFF   2048
#define M_TOK (B * S)   // 12288 tokens

typedef __attribute__((ext_vector_type(16))) __bf16       v16bf;
typedef __attribute__((ext_vector_type(8)))  float        v8f;
typedef __attribute__((ext_vector_type(4)))  unsigned int u32x4;
typedef __attribute__((ext_vector_type(4)))  int          v4i_t;

struct Frag { union { v16bf v; u32x4 u[2]; }; };

__device__ __forceinline__ v8f wmma_bf16(const Frag& a, const Frag& b, v8f c) {
  // D(f32 16x16) = A(16x32 bf16) * B(32x16 bf16) + C
  return __builtin_amdgcn_wmma_f32_16x16x32_bf16(false, a.v, false, b.v,
                                                 (short)0, c, false, false);
}

__device__ __forceinline__ float gelu_exact(float v) {
  return 0.5f * v * (1.0f + erff(v * 0.70710678118654752f));
}

// ---- CDNA5 async global->LDS path (guarded; falls back to sync staging) ----
#if defined(__has_builtin)
#if __has_builtin(__builtin_amdgcn_global_load_async_to_lds_b128)
#define HAVE_ASYNC_LDS 1
#endif
#endif

#ifdef HAVE_ASYNC_LDS
__device__ __forceinline__ void async_b128(const void* g, void* l) {
  void* gg = const_cast<void*>(g);
  __builtin_amdgcn_global_load_async_to_lds_b128(
      (__attribute__((address_space(1))) v4i_t*)gg,
      (__attribute__((address_space(3))) v4i_t*)l, 0, 0);
}
__device__ __forceinline__ void wait_async_6() {
#if __has_builtin(__builtin_amdgcn_s_wait_asynccnt)
  __builtin_amdgcn_s_wait_asynccnt(6);
#else
  asm volatile("s_wait_asynccnt 6" ::: "memory");
#endif
}
__device__ __forceinline__ void wait_async_0() {
#if __has_builtin(__builtin_amdgcn_s_wait_asynccnt)
  __builtin_amdgcn_s_wait_asynccnt(0);
#else
  asm volatile("s_wait_asynccnt 0" ::: "memory");
#endif
}
#endif

// ---------------------------------------------------------------------------
// Embedding + intensity projection: x[b,s,:] = emb[id] + inten[b,s]*iw + ib
// ---------------------------------------------------------------------------
__global__ __launch_bounds__(256) void embed_kernel(
    const int* __restrict__ ids, const float* __restrict__ inten,
    const float* __restrict__ emb, const float* __restrict__ iw,
    const float* __restrict__ ib, float* __restrict__ x)
{
  const int t  = blockIdx.x;            // token 0..12287
  const int b  = t / S, s = t % S;
  const int id = ids[t];
  const float iv = inten[b * S + s];
  const float* e = emb + (size_t)id * D;
  float* xo = x + (size_t)t * D;
  for (int d = threadIdx.x; d < D; d += 256)
    xo[d] = e[d] + iv * iw[d] + ib[d];
}

// ---------------------------------------------------------------------------
// LayerNorm: one wave per row of 512, bf16 output
// ---------------------------------------------------------------------------
__global__ __launch_bounds__(256) void ln_kernel(
    const float* __restrict__ x, const float* __restrict__ w,
    const float* __restrict__ bb, __bf16* __restrict__ out)
{
  const int wave = threadIdx.x >> 5, lane = threadIdx.x & 31;
  const int row  = blockIdx.x * 8 + wave;
  const float* xr = x + (size_t)row * D;
  float vals[16];
  float sum = 0.0f;
#pragma unroll
  for (int i = 0; i < 16; i++) { vals[i] = xr[lane + i * 32]; sum += vals[i]; }
#pragma unroll
  for (int off = 16; off >= 1; off >>= 1) sum += __shfl_xor(sum, off, 32);
  const float mean = sum * (1.0f / D);
  float var = 0.0f;
#pragma unroll
  for (int i = 0; i < 16; i++) { float d = vals[i] - mean; var += d * d; }
#pragma unroll
  for (int off = 16; off >= 1; off >>= 1) var += __shfl_xor(var, off, 32);
  const float rstd = rsqrtf(var * (1.0f / D) + 1e-5f);
  __bf16* o = out + (size_t)row * D;
#pragma unroll
  for (int i = 0; i < 16; i++) {
    int c = lane + i * 32;
    o[c] = (__bf16)((vals[i] - mean) * rstd * w[c] + bb[c]);
  }
}

// ---------------------------------------------------------------------------
// Weight convert: W fp32 [K,N] per layer -> WT bf16 [N,K] per layer
// ---------------------------------------------------------------------------
__global__ __launch_bounds__(256) void convert_wT(
    const float* __restrict__ W, __bf16* __restrict__ WT, int K, int N)
{
  const int l = blockIdx.y;
  const float* Wl  = W  + (size_t)l * K * N;
  __bf16*      WTl = WT + (size_t)l * K * N;
  const int total = K * N;
  for (int i = blockIdx.x * 256 + threadIdx.x; i < total; i += gridDim.x * 256) {
    int n = i / K, kk = i % K;
    WTl[i] = (__bf16)Wl[(size_t)kk * N + n];
  }
}

// ---------------------------------------------------------------------------
// Tiled WMMA GEMM:  C[M,N] = A[M,K](bf16) * WT[N,K]^T(bf16) + bias
// Block tile 128x256, 8 waves, wave tile 64x64 (4x4 wmma), K-step 32.
// Double-buffered LDS; tiles moved with GLOBAL_LOAD_ASYNC_TO_LDS_B128 when
// available (ASYNCcnt-tracked prefetch of tile k+1 overlapping WMMA on tile k)
// MODE 0: scatter bf16 to q/k layout [B,H,S,DK]
// MODE 1: scatter bf16 to vT layout [B,H,DK,S]
// MODE 2: residual add into fp32 X[m*N+n]
// MODE 3: exact GELU -> bf16 Out[m*N+n]
// ---------------------------------------------------------------------------
#define BMM 128
#define BNN 256
#define BKK 32
#define LDA 40   // padded bf16 row stride (80B, 16B aligned)
#define LDB 40

template<int MODE>
__global__ __launch_bounds__(256) void gemm_kernel(
    const __bf16* __restrict__ A, const __bf16* __restrict__ WT,
    const float* __restrict__ bias, float* __restrict__ X,
    __bf16* __restrict__ Out, int M, int N, int K)
{
  __shared__ __bf16 As[2][BMM * LDA];   // 2 x 10240 B
  __shared__ __bf16 Bs[2][BNN * LDB];   // 2 x 20480 B
  const int tid  = threadIdx.x;
  const int lane = tid & 31, wave = tid >> 5;
  const int wr = wave >> 2, wc = wave & 3;       // 2x4 wave grid
  const int lm = lane & 15, hi = lane >> 4;
  const int m0 = blockIdx.x * BMM, n0 = blockIdx.y * BNN;

  const int ar  = tid >> 1;              // A row for this thread
  const int ahh = (tid & 1) << 4;        // 0 or 16 (K half)
  const __bf16* Abase = A  + (size_t)(m0 + ar)  * K + ahh;
  const __bf16* Bbase = WT + (size_t)(n0 + tid) * K;

  const v8f zero = {};
  v8f acc[4][4];
#pragma unroll
  for (int i = 0; i < 4; i++)
#pragma unroll
    for (int j = 0; j < 4; j++) acc[i][j] = zero;

#ifdef HAVE_ASYNC_LDS
  // 6 async B128 ops per thread per tile (2 for A, 4 for B)
  {
    async_b128(Abase,      &As[0][ar * LDA + ahh]);
    async_b128(Abase + 8,  &As[0][ar * LDA + ahh + 8]);
    async_b128(Bbase,      &Bs[0][tid * LDB]);
    async_b128(Bbase + 8,  &Bs[0][tid * LDB + 8]);
    async_b128(Bbase + 16, &Bs[0][tid * LDB + 16]);
    async_b128(Bbase + 24, &Bs[0][tid * LDB + 24]);
  }
  int buf = 0;
  for (int k0 = 0; k0 < K; k0 += BKK) {
    const bool more = (k0 + BKK) < K;
    if (more) {   // prefetch tile k+1 into the other buffer
      const int nb = buf ^ 1;
      const __bf16* An = Abase + k0 + BKK;
      const __bf16* Bn = Bbase + k0 + BKK;
      async_b128(An,      &As[nb][ar * LDA + ahh]);
      async_b128(An + 8,  &As[nb][ar * LDA + ahh + 8]);
      async_b128(Bn,      &Bs[nb][tid * LDB]);
      async_b128(Bn + 8,  &Bs[nb][tid * LDB + 8]);
      async_b128(Bn + 16, &Bs[nb][tid * LDB + 16]);
      async_b128(Bn + 24, &Bs[nb][tid * LDB + 24]);
      wait_async_6();     // current tile's 6 ops retired; prefetch in flight
    } else {
      wait_async_0();
    }
    __syncthreads();

    Frag af[4], bf[4];
#pragma unroll
    for (int i = 0; i < 4; i++) {
      const __bf16* p = &As[buf][(wr * 64 + i * 16 + lm) * LDA + hi * 8];
      af[i].u[0] = *(const u32x4*)p;
      af[i].u[1] = *(const u32x4*)(p + 16);
    }
#pragma unroll
    for (int j = 0; j < 4; j++) {
      const __bf16* p = &Bs[buf][(wc * 64 + j * 16 + lm) * LDB + hi * 8];
      bf[j].u[0] = *(const u32x4*)p;
      bf[j].u[1] = *(const u32x4*)(p + 16);
    }
#pragma unroll
    for (int i = 0; i < 4; i++)
#pragma unroll
      for (int j = 0; j < 4; j++)
        acc[i][j] = wmma_bf16(af[i], bf[j], acc[i][j]);

    __syncthreads();      // buffer reuse safety before next prefetch
    buf ^= 1;
  }
#else
  for (int k0 = 0; k0 < K; k0 += BKK) {
    __syncthreads();
    {
      const u32x4* src = (const u32x4*)(Abase + k0);
      u32x4 d0 = src[0], d1 = src[1];
      u32x4* dst = (u32x4*)(&As[0][ar * LDA + ahh]);
      dst[0] = d0; dst[1] = d1;
    }
    {
      const u32x4* src = (const u32x4*)(Bbase + k0);
      u32x4 b0 = src[0], b1 = src[1], b2 = src[2], b3 = src[3];
      u32x4* dst = (u32x4*)(&Bs[0][tid * LDB]);
      dst[0] = b0; dst[1] = b1; dst[2] = b2; dst[3] = b3;
    }
    __syncthreads();

    Frag af[4], bf[4];
#pragma unroll
    for (int i = 0; i < 4; i++) {
      const __bf16* p = &As[0][(wr * 64 + i * 16 + lm) * LDA + hi * 8];
      af[i].u[0] = *(const u32x4*)p;
      af[i].u[1] = *(const u32x4*)(p + 16);
    }
#pragma unroll
    for (int j = 0; j < 4; j++) {
      const __bf16* p = &Bs[0][(wc * 64 + j * 16 + lm) * LDB + hi * 8];
      bf[j].u[0] = *(const u32x4*)p;
      bf[j].u[1] = *(const u32x4*)(p + 16);
    }
#pragma unroll
    for (int i = 0; i < 4; i++)
#pragma unroll
      for (int j = 0; j < 4; j++)
        acc[i][j] = wmma_bf16(af[i], bf[j], acc[i][j]);
  }
#endif

  // Epilogue. D layout: lane = column, vgpr r -> row r (+8 for high lanes)
#pragma unroll
  for (int i = 0; i < 4; i++) {
#pragma unroll
    for (int j = 0; j < 4; j++) {
      const int n = n0 + wc * 64 + j * 16 + lm;
      const float bvv = bias[n];
#pragma unroll
      for (int r = 0; r < 8; r++) {
        const int m = m0 + wr * 64 + i * 16 + r + hi * 8;
        float val = acc[i][j][r] + bvv;
        if constexpr (MODE == 2) {
          X[(size_t)m * N + n] += val;
        } else if constexpr (MODE == 3) {
          Out[(size_t)m * N + n] = (__bf16)gelu_exact(val);
        } else {
          const int b_ = m / S, s_ = m % S;
          const int h_ = n / DK, dk_ = n % DK;
          if constexpr (MODE == 0)
            Out[(((size_t)b_ * H + h_) * S + s_) * DK + dk_] = (__bf16)val;
          else
            Out[(((size_t)b_ * H + h_) * DK + dk_) * S + s_] = (__bf16)val;
        }
      }
    }
  }
}

// ---------------------------------------------------------------------------
// Attention: one wave per (b,h, 16-query tile). Scores via WMMA, softmax in
// LDS (1/sum folded into epilogue), P*V via WMMA against transposed V.
// ---------------------------------------------------------------------------
#define SCLD (S + 4)   // 388 fp32
#define PLD  (S + 8)   // 392 bf16 (784B rows, 16B aligned)

__global__ __launch_bounds__(32) void attn_kernel(
    const __bf16* __restrict__ q, const __bf16* __restrict__ k,
    const __bf16* __restrict__ vT, const int* __restrict__ ids,
    __bf16* __restrict__ out)
{
  __shared__ float  sc[16 * SCLD];
  __shared__ __bf16 p[16 * PLD];
  __shared__ float  rsum[16];
  const int lane = threadIdx.x & 31;
  const int lm = lane & 15, hi = lane >> 4;
  const int bh = blockIdx.x / 24;   // b*H + h
  const int qt = blockIdx.x % 24;   // query tile
  const int b  = bh >> 3;
  const int hh = bh & 7;

  // Q fragments (16 queries x DK=64 -> 2 K-steps), resident for all key tiles
  Frag aq[2];
  {
    const __bf16* qrow = q + ((size_t)bh * S + qt * 16 + lm) * DK;
#pragma unroll
    for (int ks = 0; ks < 2; ks++) {
      const __bf16* pp = qrow + ks * 32 + hi * 8;
      aq[ks].u[0] = *(const u32x4*)pp;
      aq[ks].u[1] = *(const u32x4*)(pp + 16);
    }
  }

  const v8f zero = {};
  for (int nt = 0; nt < 24; nt++) {          // 24 key tiles of 16
    v8f accs = zero;
    const __bf16* krow = k + ((size_t)bh * S + nt * 16 + lm) * DK;
#pragma unroll
    for (int ks = 0; ks < 2; ks++) {
      Frag bk;
      const __bf16* pp = krow + ks * 32 + hi * 8;
      bk.u[0] = *(const u32x4*)pp;
      bk.u[1] = *(const u32x4*)(pp + 16);
      accs = wmma_bf16(aq[ks], bk, accs);
    }
    const int  key = nt * 16 + lm;           // this lane's column = key index
    const bool pad = (ids[b * S + key] == 0);
#pragma unroll
    for (int r = 0; r < 8; r++) {
      float v = pad ? -1e9f : accs[r] * 0.125f;   // 1/sqrt(DK)
      sc[(r + hi * 8) * SCLD + key] = v;
    }
  }
  __syncthreads();

  {   // softmax: 2 lanes per row, 192 keys each; combine halves via shfl
    const int row = lm;
    const int c0  = hi * 192;
    float mx = -3.0e38f;
    for (int j = 0; j < 192; j++) mx = fmaxf(mx, sc[row * SCLD + c0 + j]);
    mx = fmaxf(mx, __shfl_xor(mx, 16, 32));
    float sum = 0.0f;
    for (int j = 0; j < 192; j++) {
      float e = __expf(sc[row * SCLD + c0 + j] - mx);
      p[row * PLD + c0 + j] = (__bf16)e;     // unnormalized P
      sum += e;
    }
    sum += __shfl_xor(sum, 16, 32);
    if (hi == 0) rsum[row] = sum;
  }
  __syncthreads();

  // attn = P(16xS) @ V(SxDK): 12 K-steps of 32 keys, 4 output tiles of 16
  v8f acco[4];
#pragma unroll
  for (int j = 0; j < 4; j++) acco[j] = zero;
  for (int kt = 0; kt < 12; kt++) {
    Frag ap;
    const __bf16* pp = &p[lm * PLD + kt * 32 + hi * 8];
    ap.u[0] = *(const u32x4*)pp;
    ap.u[1] = *(const u32x4*)(pp + 16);
#pragma unroll
    for (int j = 0; j < 4; j++) {
      Frag bv;
      const __bf16* vp = vT + ((size_t)bh * DK + j * 16 + lm) * S + kt * 32 + hi * 8;
      bv.u[0] = *(const u32x4*)vp;
      bv.u[1] = *(const u32x4*)(vp + 16);
      acco[j] = wmma_bf16(ap, bv, acco[j]);
    }
  }

#pragma unroll
  for (int j = 0; j < 4; j++) {
#pragma unroll
    for (int r = 0; r < 8; r++) {
      const int row = r + hi * 8;
      const float val = acco[j][r] / rsum[row];
      const size_t m = (size_t)b * S + qt * 16 + row;
      const int    n = hh * DK + j * 16 + lm;
      out[m * D + n] = (__bf16)val;
    }
  }
}

// ---------------------------------------------------------------------------
// Pool: out[b,d] = sum_s inten[b,s] * x[b,s,d]
// ---------------------------------------------------------------------------
__global__ __launch_bounds__(512) void pool_kernel(
    const float* __restrict__ inten, const float* __restrict__ x,
    float* __restrict__ out)
{
  const int b = blockIdx.x, d = threadIdx.x;
  float acc = 0.0f;
  for (int s = 0; s < S; s++)
    acc += inten[b * S + s] * x[((size_t)b * S + s) * D + d];
  out[b * D + d] = acc;
}

// ---------------------------------------------------------------------------
extern "C" void kernel_launch(void* const* d_in, const int* in_sizes, int n_in,
                              void* d_out, int out_size, void* d_ws, size_t ws_size,
                              hipStream_t stream)
{
  (void)in_sizes; (void)n_in; (void)out_size; (void)ws_size;
  const int*   ids   = (const int*)  d_in[0];
  const float* inten = (const float*)d_in[1];
  const float* emb   = (const float*)d_in[2];
  const float* iw    = (const float*)d_in[3];
  const float* ib    = (const float*)d_in[4];
  const float* Wq    = (const float*)d_in[5];
  const float* bq    = (const float*)d_in[6];
  const float* Wk    = (const float*)d_in[7];
  const float* bk    = (const float*)d_in[8];
  const float* Wv    = (const float*)d_in[9];
  const float* bv    = (const float*)d_in[10];
  const float* Wo    = (const float*)d_in[11];
  const float* bo    = (const float*)d_in[12];
  const float* ln1w  = (const float*)d_in[13];
  const float* ln1b  = (const float*)d_in[14];
  const float* ln2w  = (const float*)d_in[15];
  const float* ln2b  = (const float*)d_in[16];
  const float* W1    = (const float*)d_in[17];
  const float* b1    = (const float*)d_in[18];
  const float* W2    = (const float*)d_in[19];
  const float* b2    = (const float*)d_in[20];

  char* ws = (char*)d_ws;
  size_t off = 0;
  auto alloc = [&](size_t bytes) -> void* {
    off = (off + 255) & ~(size_t)255;
    void* p = ws + off;
    off += bytes;
    return p;
  };
  float*  x   = (float*) alloc((size_t)M_TOK * D * 4);
  __bf16* h   = (__bf16*)alloc((size_t)M_TOK * D * 2);
  __bf16* qb  = (__bf16*)alloc((size_t)M_TOK * D * 2);
  __bf16* kb  = (__bf16*)alloc((size_t)M_TOK * D * 2);
  __bf16* vTb = (__bf16*)alloc((size_t)M_TOK * D * 2);
  __bf16* ab  = (__bf16*)alloc((size_t)M_TOK * D * 2);
  __bf16* ffb = (__bf16*)alloc((size_t)M_TOK * DFF * 2);
  __bf16* WqT = (__bf16*)alloc((size_t)L * D * D * 2);
  __bf16* WkT = (__bf16*)alloc((size_t)L * D * D * 2);
  __bf16* WvT = (__bf16*)alloc((size_t)L * D * D * 2);
  __bf16* WoT = (__bf16*)alloc((size_t)L * D * D * 2);
  __bf16* W1T = (__bf16*)alloc((size_t)L * D * DFF * 2);
  __bf16* W2T = (__bf16*)alloc((size_t)L * D * DFF * 2);

  // Weight conversion (bf16 + transpose to [N,K])
  convert_wT<<<dim3(512, L), 256, 0, stream>>>(Wq, WqT, D, D);
  convert_wT<<<dim3(512, L), 256, 0, stream>>>(Wk, WkT, D, D);
  convert_wT<<<dim3(512, L), 256, 0, stream>>>(Wv, WvT, D, D);
  convert_wT<<<dim3(512, L), 256, 0, stream>>>(Wo, WoT, D, D);
  convert_wT<<<dim3(2048, L), 256, 0, stream>>>(W1, W1T, D, DFF);
  convert_wT<<<dim3(2048, L), 256, 0, stream>>>(W2, W2T, DFF, D);

  embed_kernel<<<M_TOK, 256, 0, stream>>>(ids, inten, emb, iw, ib, x);

  const dim3 g512(M_TOK / BMM, D / BNN);     // (96, 2)
  const dim3 gff (M_TOK / BMM, DFF / BNN);   // (96, 8)
  for (int l = 0; l < L; l++) {
    ln_kernel<<<M_TOK / 8, 256, 0, stream>>>(x, ln1w + l * D, ln1b + l * D, h);
    gemm_kernel<0><<<g512, 256, 0, stream>>>(h, WqT + (size_t)l * D * D, bq + l * D, nullptr, qb,  M_TOK, D, D);
    gemm_kernel<0><<<g512, 256, 0, stream>>>(h, WkT + (size_t)l * D * D, bk + l * D, nullptr, kb,  M_TOK, D, D);
    gemm_kernel<1><<<g512, 256, 0, stream>>>(h, WvT + (size_t)l * D * D, bv + l * D, nullptr, vTb, M_TOK, D, D);
    attn_kernel<<<B * H * (S / 16), 32, 0, stream>>>(qb, kb, vTb, ids, ab);
    gemm_kernel<2><<<g512, 256, 0, stream>>>(ab, WoT + (size_t)l * D * D, bo + l * D, x, nullptr, M_TOK, D, D);
    ln_kernel<<<M_TOK / 8, 256, 0, stream>>>(x, ln2w + l * D, ln2b + l * D, h);
    gemm_kernel<3><<<gff, 256, 0, stream>>>(h,  W1T + (size_t)l * D * DFF, b1 + l * DFF, nullptr, ffb, M_TOK, DFF, D);
    gemm_kernel<2><<<g512, 256, 0, stream>>>(ffb, W2T + (size_t)l * D * DFF, b2 + l * D, x, nullptr, M_TOK, D, DFF);
  }

  pool_kernel<<<B, D, 0, stream>>>(inten, x, (float*)d_out);
}
